// SparseConvNet_64785286693671
// MI455X (gfx1250) — matrix operand order repaired and checked by reading, not compiled, not run on hardware
//
#include <hip/hip_runtime.h>

// ============================================================================
// SparseConvNet classifier for MI455X (gfx1250), wave32 + WMMA.
//
// ~128 GFLOP; all intermediates fit in 192MB L2 -> minimize non-WMMA work per
// v_wmma. f16 activations + f16 *transposed* weights ([tap][n][k]) so every
// A and B fragment is two contiguous 128-bit loads (ds_load_b128 /
// global_load_b128), no per-element packing. 128-thread blocks (4 waves)
// compute 32x32 output tiles; v_wmma_f32_16x16x32_f16, f32 accumulate.
// ============================================================================

#define BN_EPS 1e-5f
#define G0 40

typedef __attribute__((ext_vector_type(16))) _Float16 v16h;
typedef __attribute__((ext_vector_type(8)))  float    v8f;

// ---------------------------------------------------------------- voxelize
__global__ void k_voxelize(const int* __restrict__ coords,
                           const float* __restrict__ feats,
                           float* __restrict__ x0,
                           unsigned char* __restrict__ mask0,
                           int B, int N) {
  int i = blockIdx.x * blockDim.x + threadIdx.x;
  if (i >= B * N) return;
  int b = i / N;
  int iz = coords[i * 3 + 0], iy = coords[i * 3 + 1], ix = coords[i * 3 + 2];
  int site = ((b * G0 + iz) * G0 + iy) * G0 + ix;
  atomicAdd(&x0[site * 3 + 0], feats[i * 3 + 0]);
  atomicAdd(&x0[site * 3 + 1], feats[i * 3 + 1]);
  atomicAdd(&x0[site * 3 + 2], feats[i * 3 + 2]);
  mask0[site] = 1;
}

// --------------------- weight convert f32 -> f16 with [t][k][n] -> [t][n][k]
__global__ void k_cvt_wT(const float* __restrict__ s, _Float16* __restrict__ d,
                         int taps, int Cin, int Cout) {
  long i = (long)blockIdx.x * blockDim.x + threadIdx.x;
  long total = (long)taps * Cin * Cout;
  if (i >= total) return;
  int  k = (int)(i % Cin);
  long r = i / Cin;
  int  n = (int)(r % Cout);
  int  t = (int)(r / Cout);
  d[i] = (_Float16)s[((long)t * Cin + k) * Cout + n];
}

// ------------------------------------------------- layer 1 (Cin=3): scalar
__global__ void k_conv1(const float* __restrict__ in, const float* __restrict__ w,
                        float* __restrict__ out, int B) {
  // in [B,40,40,40,3], w [27][3][64], out [B,40,40,40,64]
  long i = (long)blockIdx.x * blockDim.x + threadIdx.x;
  long total = (long)B * G0 * G0 * G0 * 64;
  if (i >= total) return;
  int co = (int)(i & 63);
  long s = i >> 6;
  int x = (int)(s % G0); s /= G0;
  int y = (int)(s % G0); s /= G0;
  int z = (int)(s % G0);
  int b = (int)(s / G0);
  float acc = 0.f;
  for (int t = 0; t < 27; ++t) {
    int dz = t / 9, dy = (t / 3) % 3, dx = t % 3;
    int zi = z + dz - 1, yi = y + dy - 1, xi = x + dx - 1;
    if (zi < 0 || zi >= G0 || yi < 0 || yi >= G0 || xi < 0 || xi >= G0) continue;
    const float* ip = in + (long)(((b * G0 + zi) * G0 + yi) * G0 + xi) * 3;
    const float* wp = w + (t * 3) * 64 + co;
    acc += ip[0] * wp[0] + ip[1] * wp[64] + ip[2] * wp[128];
  }
  out[i] = acc;
}

// ------------------------------------------------------------ mask dilation
__global__ void k_maskdilate(const unsigned char* __restrict__ mi,
                             unsigned char* __restrict__ mo,
                             int B, int Din, int Dout, int kdim, int stride, int pad) {
  int i = blockIdx.x * blockDim.x + threadIdx.x;
  int so = Dout * Dout * Dout;
  if (i >= B * so) return;
  int b = i / so, s = i % so;
  int xo = s % Dout, yo = (s / Dout) % Dout, zo = s / (Dout * Dout);
  unsigned char m = 0;
  int taps = kdim * kdim * kdim;
  for (int t = 0; t < taps; ++t) {
    int dz = t / (kdim * kdim), dy = (t / kdim) % kdim, dx = t % kdim;
    int zi = zo * stride + dz - pad, yi = yo * stride + dy - pad, xi = xo * stride + dx - pad;
    if (zi < 0 || zi >= Din || yi < 0 || yi >= Din || xi < 0 || xi >= Din) continue;
    m |= mi[((b * Din + zi) * Din + yi) * Din + xi];
  }
  mo[i] = m ? 1 : 0;
}

// -------------------------------- BN statistics (masked or dense), per chan
__global__ __launch_bounds__(256)
void k_bnstats(const float* __restrict__ x, const unsigned char* __restrict__ mask,
               const float* __restrict__ gamma,
               float* __restrict__ bnMean, float* __restrict__ bnScale,
               long nsites, int C) {
  __shared__ float s1[256], s2[256], sm[256];
  int c = blockIdx.x, tid = threadIdx.x;
  float a1 = 0.f, a2 = 0.f, am = 0.f;
  for (long i = tid; i < nsites; i += 256) {
    float m = mask ? (float)mask[i] : 1.f;
    float v = x[i * C + c];
    a1 += m * v; a2 += m * v * v; am += m;
  }
  s1[tid] = a1; s2[tid] = a2; sm[tid] = am;
  __syncthreads();
  for (int st = 128; st > 0; st >>= 1) {
    if (tid < st) { s1[tid] += s1[tid + st]; s2[tid] += s2[tid + st]; sm[tid] += sm[tid + st]; }
    __syncthreads();
  }
  if (tid == 0) {
    float cnt  = fmaxf(sm[0], 1.f);
    float mean = s1[0] / cnt;
    float var  = fmaxf(s2[0] / cnt - mean * mean, 0.f);
    bnMean[c]  = mean;
    bnScale[c] = gamma[c] * rsqrtf(var + BN_EPS);
  }
}

// ----------------- BN apply + ReLU (+ mask zeroing); writes f16 activations
__global__ void k_bnapply(const float* __restrict__ y, _Float16* __restrict__ xh,
                          const unsigned char* __restrict__ mask,
                          const float* __restrict__ bnMean, const float* __restrict__ bnScale,
                          const float* __restrict__ beta, long nsites, int C) {
  long i = (long)blockIdx.x * blockDim.x + threadIdx.x;
  if (i >= nsites * (long)C) return;
  long site = i / C; int c = (int)(i % C);
  float v = (y[i] - bnMean[c]) * bnScale[c] + beta[c];
  v = fmaxf(v, 0.f);
  if (mask && !mask[site]) v = 0.f;
  xh[i] = (_Float16)v;
}

// ============================================================================
// Implicit-GEMM 3D conv on WMMA, compile-time shapes.
// Block = 128 threads (4 waves). Output tile: 32 sites x 32 Cout.
// Wave w -> (mh,nh) = (w>>1, w&1) 16x16 sub-tile, f32 accumulate.
// LDS tiles: Ah[site][k], Bt[n][k] (weights pre-transposed to [tap][n][k]) so
// all fragment reads are contiguous 128-bit ds loads.
// Uniform control flow -> EXEC all-ones at every WMMA (ISA requirement).
// ============================================================================
template <int DIN, int CIN, int DOUT, int COUT, int KD, int STRIDE, int PAD>
__global__ __launch_bounds__(128)
void k_conv_wmma(const _Float16* __restrict__ in, const _Float16* __restrict__ wT,
                 float* __restrict__ out) {
  __shared__ alignas(16) _Float16 Ah[32][32];   // [site][k]
  __shared__ alignas(16) _Float16 Bt[32][32];   // [cout][k]  (transposed tile)
  const int tid  = threadIdx.x;
  const int lane = tid & 31;
  const int mh   = (tid >> 5) >> 1;     // wave row half
  const int nh   = (tid >> 5) & 1;      // wave col half
  const int m0 = blockIdx.x * 32;
  const int n0 = blockIdx.y * 32;
  const int b  = blockIdx.z;
  constexpr int sitesOut = DOUT * DOUT * DOUT;
  constexpr int taps = KD * KD * KD;
  constexpr int kcn  = CIN / 32;

  // staging role: thread copies 8 halves of row (tid>>2), segment (tid&3)
  const int arow = tid >> 2;            // A: site row   B: cout row
  const int aseg = tid & 3;             // 8-half segment
  const int site = m0 + arow;
  const bool siteValid = site < sitesOut;
  const int zo = site / (DOUT * DOUT);
  const int yo = (site / DOUT) % DOUT;
  const int xo = site % DOUT;

  v8f acc = {};
  #pragma unroll 1
  for (int tap = 0; tap < taps; ++tap) {
    const int dz = tap / (KD * KD), dy = (tap / KD) % KD, dx = tap % KD;
    const int zi = zo * STRIDE + dz - PAD;
    const int yi = yo * STRIDE + dy - PAD;
    const int xi = xo * STRIDE + dx - PAD;
    const bool inb = siteValid && zi >= 0 && zi < DIN && yi >= 0 && yi < DIN &&
                     xi >= 0 && xi < DIN;
    const _Float16* ip = in + (inb ? (long)(((b * DIN + zi) * DIN + yi) * DIN + xi) * CIN : 0);
    const _Float16* wp = wT + ((long)tap * COUT + n0 + arow) * CIN;
    #pragma unroll 1
    for (int kc = 0; kc < kcn; ++kc) {
      // ---- stage A tile (zeros for halo/padding rows -> sparse-correct)
      uint4 av = inb ? *(const uint4*)(ip + kc * 32 + aseg * 8)
                     : make_uint4(0u, 0u, 0u, 0u);
      *(uint4*)&Ah[arow][aseg * 8] = av;
      // ---- stage B tile (already [n][k] in memory): straight 128-bit copy
      *(uint4*)&Bt[arow][aseg * 8] = *(const uint4*)(wp + kc * 32 + aseg * 8);
      __syncthreads();
      // ---- per-wave fragment gather: all contiguous 16B chunks
      v16h af, bf;
      const int r  = mh * 16 + (lane & 15);
      const int kb = (lane >> 4) * 16;
      #pragma unroll
      for (int j = 0; j < 8; ++j)  af[j] = Ah[r][kb + j];          // K = base+0..7
      #pragma unroll
      for (int j = 8; j < 16; ++j) af[j] = Ah[r][kb + j];      // K = base+8..15
      const int nB = nh * 16 + (lane & 15);
      const int kB = (lane >> 4) * 16;
      #pragma unroll
      for (int j = 0; j < 16; ++j) bf[j] = Bt[nB][kB + j];         // K = kB..kB+15
      acc = __builtin_amdgcn_wmma_f32_16x16x32_f16(
          false, af, false, bf, (short)0, acc, false, false);
      __syncthreads();
    }
  }
  // ---- store: VGPR v holds row M = v + (lane<16 ? 0 : 8), col N = lane&15
  const int nC    = n0 + nh * 16 + (lane & 15);
  const int mBase = m0 + mh * 16 + (lane >> 4) * 8;
  #pragma unroll
  for (int v = 0; v < 8; ++v) {
    if (mBase + v < sitesOut)
      out[((long)b * sitesOut + mBase + v) * COUT + nC] = acc[v];
  }
}

// ------------- pack FC1 A matrix: NCDHW flatten, rows >= B zeroed
// fA[r][k] = d2h[(r*8 + (k&7))*1024 + (k>>3)]   (r<B), else 0
__global__ void k_fc1_pack(const _Float16* __restrict__ d2h,
                           _Float16* __restrict__ fA, int B) {
  int i = blockIdx.x * blockDim.x + threadIdx.x;
  if (i >= 16 * 8192) return;
  int r = i >> 13, k = i & 8191;
  _Float16 v = (_Float16)0.f;
  if (r < B) v = d2h[(r * 8 + (k & 7)) * 1024 + (k >> 3)];
  fA[i] = v;
}

// ---------------------------------------------- FC1 (8192->1024) on WMMA
// No LDS, no barriers: fragments load as contiguous 16B chunks from fA and
// the pre-transposed wl1T[n][k].
__global__ __launch_bounds__(32)
void k_fc1_wmma(const _Float16* __restrict__ fA, const _Float16* __restrict__ wl1T,
                float* __restrict__ out, int B) {
  const int lane = threadIdx.x;
  const int n0 = blockIdx.x * 16;
  const int r  = lane & 15;
  const int kb = (lane >> 4) * 16;
  const int kB = (lane >> 4) * 16;
  const _Float16* ap = fA + (long)r * 8192;
  const _Float16* bp = wl1T + (long)(n0 + (lane & 15)) * 8192;
  v8f acc = {};
  for (int kc = 0; kc < 256; ++kc) {
    const int base = kc * 32;
    v16h af, bf;
    #pragma unroll
    for (int j = 0; j < 8; ++j)  af[j] = ap[base + kb + j];
    #pragma unroll
    for (int j = 8; j < 16; ++j) af[j] = ap[base + kb + j];
    #pragma unroll
    for (int j = 0; j < 16; ++j) bf[j] = bp[base + kB + j];
    acc = __builtin_amdgcn_wmma_f32_16x16x32_f16(
        false, af, false, bf, (short)0, acc, false, false);
  }
  const int nC   = lane & 15;
  const int mAdd = (lane >> 4) * 8;
  #pragma unroll
  for (int v = 0; v < 8; ++v) {
    int bm = mAdd + v;
    if (bm < B) out[bm * 1024 + n0 + nC] = acc[v];   // bl1 cancels through BN
  }
}

// ----------------------------------- BN over batch axis (FC1) + ReLU
__global__ void k_bnfc(float* __restrict__ x, const float* __restrict__ gamma,
                       const float* __restrict__ beta, int B, int F) {
  int j = blockIdx.x * blockDim.x + threadIdx.x;
  if (j >= F) return;
  float s = 0.f, s2 = 0.f;
  for (int b = 0; b < B; ++b) { float v = x[b * F + j]; s += v; s2 += v * v; }
  float mean = s / B;
  float var  = fmaxf(s2 / B - mean * mean, 0.f);
  float sc   = gamma[j] * rsqrtf(var + BN_EPS);
  for (int b = 0; b < B; ++b)
    x[b * F + j] = fmaxf((x[b * F + j] - mean) * sc + beta[j], 0.f);
}

// ------------------------------------------------ FC2 (1024->40) + bias
__global__ void k_fc2(const float* __restrict__ x, const float* __restrict__ w,
                      const float* __restrict__ bias, float* __restrict__ out,
                      int B, int K, int O) {
  int i = blockIdx.x * blockDim.x + threadIdx.x;
  if (i >= B * O) return;
  int b = i / O, o = i % O;
  float acc = bias[o];
  for (int k = 0; k < K; ++k) acc += x[b * K + k] * w[k * O + o];
  out[i] = acc;
}

// ============================================================================
extern "C" void kernel_launch(void* const* d_in, const int* in_sizes, int n_in,
                              void* d_out, int out_size, void* d_ws, size_t ws_size,
                              hipStream_t stream) {
  (void)in_sizes; (void)n_in; (void)out_size; (void)ws_size;
  const int B = 8, N = 6144;

  // ---- inputs (setup_inputs dict order) ----
  const int*   coords = (const int*)  d_in[0];
  const float* feats  = (const float*)d_in[1];
  const float* w1 = (const float*)d_in[2],  *g1 = (const float*)d_in[3],  *b1 = (const float*)d_in[4];
  const float* w2 = (const float*)d_in[5],  *g2 = (const float*)d_in[6],  *b2 = (const float*)d_in[7];
  const float* w3 = (const float*)d_in[8],  *g3 = (const float*)d_in[9],  *b3 = (const float*)d_in[10];
  const float* w4 = (const float*)d_in[11], *g4 = (const float*)d_in[12], *b4 = (const float*)d_in[13];
  const float* w5 = (const float*)d_in[14], *g5 = (const float*)d_in[15], *b5 = (const float*)d_in[16];
  const float* wd1 = (const float*)d_in[17];                      // bd1 (d_in[18]) cancels in BN
  const float* gd1 = (const float*)d_in[19], *betad1 = (const float*)d_in[20];
  const float* wd2 = (const float*)d_in[21];                      // bd2 cancels in BN
  const float* gd2 = (const float*)d_in[23], *betad2 = (const float*)d_in[24];
  const float* wl1 = (const float*)d_in[25];                      // bl1 cancels in BN
  const float* gl1 = (const float*)d_in[27], *betal1 = (const float*)d_in[28];
  const float* wl2 = (const float*)d_in[29], *bl2 = (const float*)d_in[30];
  float* outp = (float*)d_out;

  // ---- workspace carve-up ----
  char* ws = (char*)d_ws;
  size_t off = 0;
  auto carve = [&](size_t bytes) {
    char* p = ws + off;
    off = (off + bytes + 255) & ~(size_t)255;
    return (void*)p;
  };
  float*         x0   = (float*)        carve((size_t)B * 64000 * 3 * 4);
  unsigned char* msk0 = (unsigned char*)carve((size_t)B * 64000);
  unsigned char* m1   = (unsigned char*)carve((size_t)B * 64000);
  unsigned char* m2   = (unsigned char*)carve((size_t)B * 8000);
  unsigned char* m3   = (unsigned char*)carve((size_t)B * 8000);
  unsigned char* m4   = (unsigned char*)carve((size_t)B * 1000);
  unsigned char* m5   = (unsigned char*)carve((size_t)B * 1000);
  float*         y    = (float*)        carve((size_t)B * 64000 * 64 * 4);  // pre-BN, reused all layers
  _Float16*      hA   = (_Float16*)     carve((size_t)B * 64000 * 64 * 2);  // f16 act ping
  _Float16*      hB   = (_Float16*)     carve((size_t)B * 8000 * 64 * 2);   // f16 act pong
  _Float16*      fA   = (_Float16*)     carve((size_t)16 * 8192 * 2);       // packed FC1 A
  float*         fc1  = (float*)        carve((size_t)B * 1024 * 4);
  float*         bnMean  = (float*)     carve(1024 * 4);
  float*         bnScale = (float*)     carve(1024 * 4);
  _Float16* w2h  = (_Float16*)carve((size_t)27 * 64 * 64 * 2);     // all wT: [tap][n][k]
  _Float16* w3h  = (_Float16*)carve((size_t)27 * 64 * 128 * 2);
  _Float16* w4h  = (_Float16*)carve((size_t)27 * 128 * 256 * 2);
  _Float16* w5h  = (_Float16*)carve((size_t)27 * 256 * 512 * 2);
  _Float16* wd1h = (_Float16*)carve((size_t)8 * 512 * 1024 * 2);
  _Float16* wd2h = (_Float16*)carve((size_t)8 * 1024 * 1024 * 2);
  _Float16* wl1h = (_Float16*)carve((size_t)8192 * 1024 * 2);

  // ---- one-time weight conversions f32 -> f16, transposed to [tap][n][k] ----
  auto cvtT = [&](const float* s, _Float16* d, int taps, int ci, int co) {
    long n = (long)taps * ci * co;
    k_cvt_wT<<<(int)((n + 255) / 256), 256, 0, stream>>>(s, d, taps, ci, co);
  };
  cvtT(w2, w2h, 27, 64, 64);
  cvtT(w3, w3h, 27, 64, 128);
  cvtT(w4, w4h, 27, 128, 256);
  cvtT(w5, w5h, 27, 256, 512);
  cvtT(wd1, wd1h, 8, 512, 1024);
  cvtT(wd2, wd2h, 8, 1024, 1024);
  cvtT(wl1, wl1h, 1, 8192, 1024);

  // ---- voxelize (scatter-add) ----
  hipMemsetAsync(x0,   0, (size_t)B * 64000 * 3 * 4, stream);
  hipMemsetAsync(msk0, 0, (size_t)B * 64000,         stream);
  k_voxelize<<<(B * N + 255) / 256, 256, 0, stream>>>(coords, feats, x0, msk0, B, N);

  // ---- block 1: 40^3, 3->64, s1 (scalar; Cin=3 is a poor WMMA fit) ----
  k_conv1<<<(int)(((long)B * 64000 * 64 + 255) / 256), 256, 0, stream>>>(x0, w1, y, B);
  k_maskdilate<<<(B * 64000 + 255) / 256, 256, 0, stream>>>(msk0, m1, B, 40, 40, 3, 1, 1);
  k_bnstats<<<64, 256, 0, stream>>>(y, m1, g1, bnMean, bnScale, (long)B * 64000, 64);
  k_bnapply<<<(int)(((long)B * 64000 * 64 + 255) / 256), 256, 0, stream>>>(
      y, hA, m1, bnMean, bnScale, b1, (long)B * 64000, 64);

  // ---- block 2: 40->20, 64->64, s2 (WMMA) ----
  k_conv_wmma<40, 64, 20, 64, 3, 2, 1><<<dim3(250, 2, B), 128, 0, stream>>>(hA, w2h, y);
  k_maskdilate<<<(B * 8000 + 255) / 256, 256, 0, stream>>>(m1, m2, B, 40, 20, 3, 2, 1);
  k_bnstats<<<64, 256, 0, stream>>>(y, m2, g2, bnMean, bnScale, (long)B * 8000, 64);
  k_bnapply<<<(int)(((long)B * 8000 * 64 + 255) / 256), 256, 0, stream>>>(
      y, hB, m2, bnMean, bnScale, b2, (long)B * 8000, 64);

  // ---- block 3: 20^3, 64->128, s1 (WMMA) ----
  k_conv_wmma<20, 64, 20, 128, 3, 1, 1><<<dim3(250, 4, B), 128, 0, stream>>>(hB, w3h, y);
  k_maskdilate<<<(B * 8000 + 255) / 256, 256, 0, stream>>>(m2, m3, B, 20, 20, 3, 1, 1);
  k_bnstats<<<128, 256, 0, stream>>>(y, m3, g3, bnMean, bnScale, (long)B * 8000, 128);
  k_bnapply<<<(int)(((long)B * 8000 * 128 + 255) / 256), 256, 0, stream>>>(
      y, hA, m3, bnMean, bnScale, b3, (long)B * 8000, 128);

  // ---- block 4: 20->10, 128->256, s2 (WMMA) ----
  k_conv_wmma<20, 128, 10, 256, 3, 2, 1><<<dim3(32, 8, B), 128, 0, stream>>>(hA, w4h, y);
  k_maskdilate<<<(B * 1000 + 255) / 256, 256, 0, stream>>>(m3, m4, B, 20, 10, 3, 2, 1);
  k_bnstats<<<256, 256, 0, stream>>>(y, m4, g4, bnMean, bnScale, (long)B * 1000, 256);
  k_bnapply<<<(int)(((long)B * 1000 * 256 + 255) / 256), 256, 0, stream>>>(
      y, hB, m4, bnMean, bnScale, b4, (long)B * 1000, 256);

  // ---- block 5: 10^3, 256->512, s1 (WMMA) ----
  k_conv_wmma<10, 256, 10, 512, 3, 1, 1><<<dim3(32, 16, B), 128, 0, stream>>>(hB, w5h, y);
  k_maskdilate<<<(B * 1000 + 255) / 256, 256, 0, stream>>>(m4, m5, B, 10, 10, 3, 1, 1);
  k_bnstats<<<512, 256, 0, stream>>>(y, m5, g5, bnMean, bnScale, (long)B * 1000, 512);
  k_bnapply<<<(int)(((long)B * 1000 * 512 + 255) / 256), 256, 0, stream>>>(
      y, hA, m5, bnMean, bnScale, b5, (long)B * 1000, 512);

  // ---- dense 1: 10->5, 512->1024, k2 s2 p0 (WMMA); bias bd1 cancels in BN
  k_conv_wmma<10, 512, 5, 1024, 2, 2, 0><<<dim3(4, 32, B), 128, 0, stream>>>(hA, wd1h, y);
  k_bnstats<<<1024, 256, 0, stream>>>(y, nullptr, gd1, bnMean, bnScale, (long)B * 125, 1024);
  k_bnapply<<<(int)(((long)B * 125 * 1024 + 255) / 256), 256, 0, stream>>>(
      y, hB, nullptr, bnMean, bnScale, betad1, (long)B * 125, 1024);

  // ---- dense 2: 5->2, 1024->1024, k2 s2 p0 (WMMA); bias bd2 cancels in BN
  k_conv_wmma<5, 1024, 2, 1024, 2, 2, 0><<<dim3(1, 32, B), 128, 0, stream>>>(hB, wd2h, y);
  k_bnstats<<<1024, 256, 0, stream>>>(y, nullptr, gd2, bnMean, bnScale, (long)B * 8, 1024);
  k_bnapply<<<(int)(((long)B * 8 * 1024 + 255) / 256), 256, 0, stream>>>(
      y, hA, nullptr, bnMean, bnScale, betad2, (long)B * 8, 1024);

  // ---- FC head ----
  k_fc1_pack<<<(16 * 8192 + 255) / 256, 256, 0, stream>>>(hA, fA, B);
  k_fc1_wmma<<<64, 32, 0, stream>>>(fA, wl1h, fc1, B);
  k_bnfc<<<(1024 + 255) / 256, 256, 0, stream>>>(fc1, gl1, betal1, B, 1024);
  k_fc2<<<(B * 40 + 255) / 256, 256, 0, stream>>>(fc1, wl2, bl2, outp, B, 1024, 40);
}
